// FaceAttention3_16217796509902
// MI455X (gfx1250) — matrix operand order repaired and checked by reading, not compiled, not run on hardware
//
#include <hip/hip_runtime.h>
#include <hip/hip_bf16.h>
#include <math.h>

typedef __bf16 bf16_t;
typedef __attribute__((ext_vector_type(16))) __bf16 v16bf;
typedef __attribute__((ext_vector_type(8)))  float  v8f;

#define Bq   32
#define Sq   512
#define Mq   8
#define DF   512
#define DIM1 2048
#define H    512
#define ROWS (Bq * Sq)          // 16384
#define EPSq 1e-5f

__device__ __forceinline__ float fast_tanh(float x) {
#if __has_builtin(__builtin_amdgcn_tanhf)
    return __builtin_amdgcn_tanhf(x);
#elif __has_builtin(__builtin_amdgcn_tanh_f32)
    return __builtin_amdgcn_tanh_f32(x);
#else
    return tanhf(x);
#endif
}

// ---------------------------------------------------------------------------
// Kernel 1: convert W1[:, :2048] (row-major [512][2560]) to bf16 [512][2048]
// ---------------------------------------------------------------------------
__global__ __launch_bounds__(256) void conv_w1a_kernel(
    const float* __restrict__ W1, bf16_t* __restrict__ w1a) {
    int idx = blockIdx.x * 256 + threadIdx.x;        // 0 .. 512*2048-1
    int h = idx >> 11;
    int d = idx & 2047;
    w1a[idx] = (bf16_t)W1[(size_t)h * (DIM1 + H) + d];
}

// ---------------------------------------------------------------------------
// Kernel 2: part_mem[b,m,h] = b1[h] + dot(memory[b,m,:], W1[h, 2048:2560])
// ---------------------------------------------------------------------------
__global__ __launch_bounds__(256) void part_mem_kernel(
    const float* __restrict__ memory, const float* __restrict__ W1,
    const float* __restrict__ b1, float* __restrict__ part_mem) {
    int idx = blockIdx.x * 256 + threadIdx.x;        // 0 .. 131071
    int h  = idx & 511;
    int bm = idx >> 9;
    const float* mem = memory + (size_t)bm * H;
    const float* w   = W1 + (size_t)h * (DIM1 + H) + DIM1;
    float acc = b1[h];
    #pragma unroll 4
    for (int d = 0; d < H; d += 4) {
        acc = fmaf(mem[d + 0], w[d + 0], acc);
        acc = fmaf(mem[d + 1], w[d + 1], acc);
        acc = fmaf(mem[d + 2], w[d + 2], acc);
        acc = fmaf(mem[d + 3], w[d + 3], acc);
    }
    part_mem[idx] = acc;
}

// ---------------------------------------------------------------------------
// Kernel 3: WMMA GEMM  part_base[16384][512] = base[16384][2048] @ W1a^T
// base = concat(face, fc, img, broadcast(label)) built on the fly in bf16.
// One wave -> 16x64 output strip, 4 accumulators, K in steps of 32.
// ---------------------------------------------------------------------------
__global__ __launch_bounds__(256) void gemm_base_kernel(
    const float* __restrict__ face, const float* __restrict__ fc,
    const float* __restrict__ img,  const float* __restrict__ label,
    const bf16_t* __restrict__ w1a, float* __restrict__ part_base) {

    const int wave   = (blockIdx.x * 256 + threadIdx.x) >> 5;  // 0..8191
    const int lane   = threadIdx.x & 31;
    const int laneLo = lane & 15;
    const int hi     = lane >> 4;
    const int mTile  = wave >> 3;            // 0..1023
    const int n0     = (wave & 7) * 64;      // 0..448

    const int row = mTile * 16 + laneLo;     // A-matrix row for this lane
    const int b   = row >> 9;                // row / 512

    const float* srcs[4] = {
        face  + (size_t)row * DF,
        fc    + (size_t)row * DF,
        img   + (size_t)row * DF,
        label + (size_t)b   * DF             // broadcast over s
    };

    // B base: lane laneLo covers columns n0+laneLo+{0,16,32,48}; hi selects K half
    const bf16_t* bbase = w1a + (size_t)(n0 + laneLo) * DIM1 + hi * 16;

    v8f acc0 = {}, acc1 = {}, acc2 = {}, acc3 = {};

    #pragma unroll
    for (int seg = 0; seg < 4; ++seg) {
        const float* sp = srcs[seg];
        for (int kk = 0; kk < DF; kk += 32) {
            const int k0 = seg * DF + kk;
            // A fragment: lane<16 -> K {0..15}; lane>=16 -> K {16..31}
            const float4* g0 = (const float4*)(sp + kk + hi * 16);
            const float4* g1 = (const float4*)(sp + kk + hi * 16 + 8);
            float4 x0 = g0[0], x1 = g0[1], x2 = g1[0], x3 = g1[1];
            v16bf a;
            a[0]  = (bf16_t)x0.x; a[1]  = (bf16_t)x0.y; a[2]  = (bf16_t)x0.z; a[3]  = (bf16_t)x0.w;
            a[4]  = (bf16_t)x1.x; a[5]  = (bf16_t)x1.y; a[6]  = (bf16_t)x1.z; a[7]  = (bf16_t)x1.w;
            a[8]  = (bf16_t)x2.x; a[9]  = (bf16_t)x2.y; a[10] = (bf16_t)x2.z; a[11] = (bf16_t)x2.w;
            a[12] = (bf16_t)x3.x; a[13] = (bf16_t)x3.y; a[14] = (bf16_t)x3.z; a[15] = (bf16_t)x3.w;

            // B fragments: 16 contiguous bf16 per lane per n-subtile
            const bf16_t* bp = bbase + k0;
            v16bf b0v = *(const v16bf*)(bp);
            v16bf b1v = *(const v16bf*)(bp + 16 * DIM1);
            v16bf b2v = *(const v16bf*)(bp + 32 * DIM1);
            v16bf b3v = *(const v16bf*)(bp + 48 * DIM1);

            acc0 = __builtin_amdgcn_wmma_f32_16x16x32_bf16(false, a, false, b0v, (short)0, acc0, false, false);
            acc1 = __builtin_amdgcn_wmma_f32_16x16x32_bf16(false, a, false, b1v, (short)0, acc1, false, false);
            acc2 = __builtin_amdgcn_wmma_f32_16x16x32_bf16(false, a, false, b2v, (short)0, acc2, false, false);
            acc3 = __builtin_amdgcn_wmma_f32_16x16x32_bf16(false, a, false, b3v, (short)0, acc3, false, false);
        }
    }

    // D layout: VGPR i, lanes 0-15: M=i, N=laneLo; lanes 16-31: M=8+i
    const int rbase = mTile * 16 + hi * 8;
    #pragma unroll
    for (int i = 0; i < 8; ++i) {
        float* out = part_base + (size_t)(rbase + i) * H + n0 + laneLo;
        out[0]  = acc0[i];
        out[16] = acc1[i];
        out[32] = acc2[i];
        out[48] = acc3[i];
    }
}

// ---------------------------------------------------------------------------
// Kernel 4: e[b,m,s] = sum_h tanh(part_base[b,s,h] + part_mem[b,m,h]) * w2[h]
// one wave per (b,s); part_base row read once, reused for all 8 memories.
// ---------------------------------------------------------------------------
__global__ __launch_bounds__(256) void e_kernel(
    const float* __restrict__ part_base, const float* __restrict__ part_mem,
    const float* __restrict__ w2, float* __restrict__ e) {
    const int wave = (blockIdx.x * 256 + threadIdx.x) >> 5;  // 0..16383 = b*512+s
    const int lane = threadIdx.x & 31;
    const int b = wave >> 9;
    const int s = wave & 511;

    const float* pb = part_base + (size_t)wave * H;
    float pbr[16], w2r[16];
    #pragma unroll
    for (int i = 0; i < 16; ++i) {
        pbr[i] = pb[lane + i * 32];
        w2r[i] = w2[lane + i * 32];
    }
    #pragma unroll
    for (int m = 0; m < Mq; ++m) {
        const float* pm = part_mem + (size_t)(b * Mq + m) * H;
        float sum = 0.f;
        #pragma unroll
        for (int i = 0; i < 16; ++i) {
            float t = fast_tanh(pbr[i] + pm[lane + i * 32]);
            sum = fmaf(t, w2r[i], sum);
        }
        #pragma unroll
        for (int off = 16; off > 0; off >>= 1)
            sum += __shfl_xor(sum, off, 32);
        if (lane == 0)
            e[(size_t)b * (Mq * Sq) + m * Sq + s] = sum;
    }
}

// ---------------------------------------------------------------------------
// Kernel 5: per-batch softmax over 4096, mask by face_mask[b,s], renormalize.
// alpha written directly into d_out.
// ---------------------------------------------------------------------------
__global__ __launch_bounds__(512) void softmax_kernel(
    const float* __restrict__ e, const float* __restrict__ face_mask,
    float* __restrict__ alpha) {
    const int b = blockIdx.x;
    const int t = threadIdx.x;
    __shared__ float red[512];

    const float* eb = e + (size_t)b * (Mq * Sq);
    float v[Mq];
    float mx = -INFINITY;
    #pragma unroll
    for (int i = 0; i < Mq; ++i) { v[i] = eb[i * Sq + t]; mx = fmaxf(mx, v[i]); }

    red[t] = mx; __syncthreads();
    for (int off = 256; off > 0; off >>= 1) {
        if (t < off) red[t] = fmaxf(red[t], red[t + off]);
        __syncthreads();
    }
    mx = red[0]; __syncthreads();

    float p[Mq]; float sum = 0.f;
    #pragma unroll
    for (int i = 0; i < Mq; ++i) { p[i] = __expf(v[i] - mx); sum += p[i]; }

    // element (m=i, s=t): mask depends only on s -> same mask for all 8 values
    const float mask = face_mask[b * Sq + t];

    red[t] = sum; __syncthreads();
    for (int off = 256; off > 0; off >>= 1) {
        if (t < off) red[t] += red[t + off];
        __syncthreads();
    }
    const float Z = red[0]; __syncthreads();

    red[t] = sum * mask; __syncthreads();
    for (int off = 256; off > 0; off >>= 1) {
        if (t < off) red[t] += red[t + off];
        __syncthreads();
    }
    const float PM = red[0];

    // out = (p/Z)*mask / (PM/Z + EPS) = p*mask / (PM + Z*EPS)
    const float inv = 1.0f / (PM + Z * EPSq);
    float* ab = alpha + (size_t)b * (Mq * Sq);
    #pragma unroll
    for (int i = 0; i < Mq; ++i)
        ab[i * Sq + t] = p[i] * mask * inv;
}

// ---------------------------------------------------------------------------
// Kernel 6: context[b,d] = sum_s (sum_m alpha[b,m,s]) * face[b,s,d]
// ---------------------------------------------------------------------------
__global__ __launch_bounds__(512) void context_kernel(
    const float* __restrict__ alpha, const float* __restrict__ face,
    float* __restrict__ context) {
    const int b = blockIdx.x;
    const int t = threadIdx.x;
    __shared__ float w[512];

    const float* ab = alpha + (size_t)b * (Mq * Sq);
    float ws = 0.f;
    #pragma unroll
    for (int m = 0; m < Mq; ++m) ws += ab[m * Sq + t];
    w[t] = ws;
    __syncthreads();

    const float* fb = face + (size_t)b * Sq * DF;
    float acc = 0.f;
    for (int s = 0; s < Sq; ++s)
        acc = fmaf(w[s], fb[(size_t)s * DF + t], acc);
    context[(size_t)b * DF + t] = acc;
}

// ---------------------------------------------------------------------------
extern "C" void kernel_launch(void* const* d_in, const int* in_sizes, int n_in,
                              void* d_out, int out_size, void* d_ws, size_t ws_size,
                              hipStream_t stream) {
    // setup_inputs order: fc, img, label, memory, face, face_mask, W1, b1, w2
    const float* fc        = (const float*)d_in[0];
    const float* img       = (const float*)d_in[1];
    const float* label     = (const float*)d_in[2];
    const float* memory    = (const float*)d_in[3];
    const float* face      = (const float*)d_in[4];
    const float* face_mask = (const float*)d_in[5];
    const float* W1        = (const float*)d_in[6];
    const float* b1        = (const float*)d_in[7];
    const float* w2        = (const float*)d_in[8];

    float* out_alpha   = (float*)d_out;                       // [32, 4096]
    float* out_context = (float*)d_out + (size_t)Bq * Mq * Sq; // [32, 512]

    // workspace layout (all offsets 256B-aligned)
    char* ws = (char*)d_ws;
    float*  part_base = (float*)ws;                                   // 33,554,432 B
    bf16_t* w1a       = (bf16_t*)(ws + 33554432);                     //  2,097,152 B
    float*  part_mem  = (float*)(ws + 33554432 + 2097152);            //    524,288 B
    float*  e_buf     = (float*)(ws + 33554432 + 2097152 + 524288);   //    524,288 B

    conv_w1a_kernel<<<(H * DIM1) / 256, 256, 0, stream>>>(W1, w1a);
    part_mem_kernel<<<(Bq * Mq * H) / 256, 256, 0, stream>>>(memory, W1, b1, part_mem);
    gemm_base_kernel<<<1024, 256, 0, stream>>>(face, fc, img, label, w1a, part_base);
    e_kernel<<<(ROWS * 32) / 256, 256, 0, stream>>>(part_base, part_mem, w2, e_buf);
    softmax_kernel<<<Bq, 512, 0, stream>>>(e_buf, face_mask, out_alpha);
    context_kernel<<<Bq, 512, 0, stream>>>(out_alpha, face, out_context);
}
